// MFVILinear_33492154974381
// MI455X (gfx1250) — matrix operand order repaired and checked by reading, not compiled, not run on hardware
//
#include <hip/hip_runtime.h>

#define EPSF 1e-5f
#define B_ROWS 1024
#define IN_DIM 4096
#define OUT_DIM 4096

// Block tile: 128(M) x 256(N), 8 waves (2x4), each wave 64x64.
#define BM 128
#define BN 256
#define BK 32
#define TILE_PITCH 80                 // 64B row + 16B TDM pad -> conflict-free LDS
#define A_BYTES (BM * TILE_PITCH)     // 10240
#define B_BYTES (BN * TILE_PITCH)     // 20480
#define BUF_BYTES (A_BYTES + B_BYTES) // 30720 ; x2 buffers = 61440 B LDS

typedef __attribute__((ext_vector_type(16))) __bf16 v16bf;
typedef __attribute__((ext_vector_type(8)))  __bf16 v8bf;
typedef __attribute__((ext_vector_type(8)))  float  v8f;
typedef __attribute__((ext_vector_type(4)))  unsigned int u32x4;
typedef __attribute__((ext_vector_type(8)))  int i32x8;
typedef __attribute__((ext_vector_type(4)))  int i32x4;

__device__ __forceinline__ unsigned short f32_to_bf16_bits(float f) {
    union { float f; unsigned int u; } v; v.f = f;
    unsigned int r = v.u + 0x7FFFu + ((v.u >> 16) & 1u);   // round-to-nearest-even
    return (unsigned short)(r >> 16);
}

// ---- TDM: 2D tile (tile_d0 contiguous elems x tile_d1 rows) of a 2D bf16 tensor -> LDS ----
// D# packing per CDNA5 ISA. data_size=1 (2 bytes).
// pad_enable: +4 DWORDs every 16 DWORDs -> LDS row pitch 80B.
__device__ __forceinline__ void tdm_load_2d_bf16(unsigned lds_addr, const void* gaddr,
                                                 unsigned tensor_d0, unsigned tensor_d1,
                                                 unsigned tile_d0, unsigned tile_d1,
                                                 unsigned stride_d0)
{
    unsigned long long ga = (unsigned long long)(__SIZE_TYPE__)gaddr;
    u32x4 g0;
    g0[0] = 1u;                                             // count=1, user descriptor
    g0[1] = lds_addr;                                       // LDS byte address
    g0[2] = (unsigned)ga;                                   // global_addr[31:0]
    g0[3] = (unsigned)((ga >> 32) & 0x01FFFFFFu) | (2u << 30); // addr[56:32] | type=2
    i32x8 g1;
    g1[0] = (int)((1u << 16)        // data_size = 2 bytes
                | (1u << 20)        // pad_enable
                | (3u << 22)        // pad_interval: 16 DWORDs (64B)
                | (3u << 25));      // pad_amount: 4 DWORDs (16B)
    g1[1] = (int)((tensor_d0 & 0xFFFFu) << 16);                              // dim0[15:0]
    g1[2] = (int)(((tensor_d0 >> 16) & 0xFFFFu) | ((tensor_d1 & 0xFFFFu) << 16));
    g1[3] = (int)(((tensor_d1 >> 16) & 0xFFFFu) | ((tile_d0 & 0xFFFFu) << 16));
    g1[4] = (int)(tile_d1 & 0xFFFFu);                                        // tile_dim1
    g1[5] = (int)stride_d0;                                                  // dim0_stride
    g1[6] = 0;
    g1[7] = 0;
    i32x4 z4 = {0, 0, 0, 0};
#if defined(__clang_major__) && (__clang_major__ >= 23)
    i32x8 z8 = {0, 0, 0, 0, 0, 0, 0, 0};
    __builtin_amdgcn_tensor_load_to_lds(g0, g1, z4, z4, z8, 0);
#else
    __builtin_amdgcn_tensor_load_to_lds(g0, g1, z4, z4, 0);
#endif
}

// ---------------- Pass 1a: W = Wmean + noise_w * max(exp(Wstd), EPS)  -> bf16 ----------------
__global__ __launch_bounds__(256)
void mat_weight_kernel(const float* __restrict__ wm, const float* __restrict__ wsp,
                       const float* __restrict__ nw, unsigned short* __restrict__ wout)
{
    const size_t i = ((size_t)blockIdx.x * 256 + threadIdx.x) * 4;
    const float4 m = *(const float4*)(wm  + i);
    const float4 s = *(const float4*)(wsp + i);
    const float4 n = *(const float4*)(nw  + i);
    ushort4 o;
    o.x = f32_to_bf16_bits(fmaf(n.x, fmaxf(__expf(s.x), EPSF), m.x));
    o.y = f32_to_bf16_bits(fmaf(n.y, fmaxf(__expf(s.y), EPSF), m.y));
    o.z = f32_to_bf16_bits(fmaf(n.z, fmaxf(__expf(s.z), EPSF), m.z));
    o.w = f32_to_bf16_bits(fmaf(n.w, fmaxf(__expf(s.w), EPSF), m.w));
    *(ushort4*)(wout + i) = o;
}

// ---------------- Pass 1b: x (f32) -> bf16 ----------------
__global__ __launch_bounds__(256)
void mat_x_kernel(const float* __restrict__ x, unsigned short* __restrict__ xout)
{
    const size_t i = ((size_t)blockIdx.x * 256 + threadIdx.x) * 4;
    const float4 v = *(const float4*)(x + i);
    ushort4 o;
    o.x = f32_to_bf16_bits(v.x);
    o.y = f32_to_bf16_bits(v.y);
    o.z = f32_to_bf16_bits(v.z);
    o.w = f32_to_bf16_bits(v.w);
    *(ushort4*)(xout + i) = o;
}

// ---------------- Pass 1c: bias ----------------
__global__ __launch_bounds__(256)
void mat_bias_kernel(const float* __restrict__ bm, const float* __restrict__ bsp,
                     const float* __restrict__ nb, float* __restrict__ bout)
{
    const int o = blockIdx.x * 256 + threadIdx.x;
    bout[o] = fmaf(nb[o], fmaxf(__expf(bsp[o]), EPSF), bm[o]);
}

// ---------------- Pass 2: TDM->LDS double-buffered bf16 WMMA GEMM ----------------
__global__ __launch_bounds__(256)
void gemm_wmma_kernel(const __bf16* __restrict__ Xb, const __bf16* __restrict__ Wb,
                      const float* __restrict__ bias, float* __restrict__ C)
{
    __shared__ __align__(1024) unsigned char smem[2 * BUF_BYTES];

    const int lane = threadIdx.x & 31;
    const int wave = threadIdx.x >> 5;
    const int wmL = (wave >> 2) * 64;              // wave M offset within block
    const int wnL = (wave &  3) * 64;              // wave N offset within block
    const int mBlk = blockIdx.y * BM;
    const int nBlk = blockIdx.x * BN;

    const int lm = lane & 15;                       // row (A) / col (B,D) within 16
    const int aByte = ((lane >> 4) << 4);           // A: lanes 16-31 start at K+8 -> +16B
    const int bByte = ((lane >> 4) << 4);           // B: lanes 16-31 start at K+8 -> +16B

    const unsigned ldsBase = (unsigned)(__SIZE_TYPE__)&smem[0];

    v8f acc[4][4] = {};

    const int KSTEPS = IN_DIM / BK;                 // 128

    // Prologue: stage k=0 tiles into buffer 0 (one wave drives the TDM).
    if (wave == 0) {
        tdm_load_2d_bf16(ldsBase,           Xb + (size_t)mBlk * IN_DIM,
                         IN_DIM, B_ROWS,  BK, BM, IN_DIM);
        tdm_load_2d_bf16(ldsBase + A_BYTES, Wb + (size_t)nBlk * IN_DIM,
                         IN_DIM, OUT_DIM, BK, BN, IN_DIM);
    }

    for (int ks = 0; ks < KSTEPS; ++ks) {
        const int cur = ks & 1;
        if (wave == 0) {
            if (ks + 1 < KSTEPS) {
                const unsigned nb_ = ldsBase + (unsigned)((cur ^ 1) * BUF_BYTES);
                const int kn = (ks + 1) * BK;
                tdm_load_2d_bf16(nb_,           Xb + (size_t)mBlk * IN_DIM + kn,
                                 IN_DIM, B_ROWS,  BK, BM, IN_DIM);
                tdm_load_2d_bf16(nb_ + A_BYTES, Wb + (size_t)nBlk * IN_DIM + kn,
                                 IN_DIM, OUT_DIM, BK, BN, IN_DIM);
                __builtin_amdgcn_s_wait_tensorcnt(2);   // current buffer's pair done
            } else {
                __builtin_amdgcn_s_wait_tensorcnt(0);
            }
        }
        __syncthreads();

        const unsigned char* buf = smem + cur * BUF_BYTES;
        const unsigned char* Ab  = buf;
        const unsigned char* Bb  = buf + A_BYTES;

        v16bf bfrag[4];
        #pragma unroll
        for (int ni = 0; ni < 4; ++ni) {
            const unsigned char* p = Bb + (wnL + ni * 16 + lm) * TILE_PITCH + bByte;
            v8bf lo = *(const v8bf*)p;
            v8bf hi = *(const v8bf*)(p + 32);
            bfrag[ni] = __builtin_shufflevector(lo, hi, 0,1,2,3,4,5,6,7,
                                                         8,9,10,11,12,13,14,15);
        }
        #pragma unroll
        for (int mi = 0; mi < 4; ++mi) {
            const unsigned char* p = Ab + (wmL + mi * 16 + lm) * TILE_PITCH + aByte;
            v8bf lo = *(const v8bf*)p;          // K = aK .. aK+7
            v8bf hi = *(const v8bf*)(p + 32);   // K = aK+16 .. aK+23
            v16bf a = __builtin_shufflevector(lo, hi, 0,1,2,3,4,5,6,7,
                                                       8,9,10,11,12,13,14,15);
            #pragma unroll
            for (int ni = 0; ni < 4; ++ni)
                acc[mi][ni] = __builtin_amdgcn_wmma_f32_16x16x32_bf16(
                    false, a, false, bfrag[ni], (short)0, acc[mi][ni], false, false);
        }
        __syncthreads();   // all reads of buf[cur] done before it is refilled
    }

    // Epilogue: bias add + store (D layout: VGPR r -> row r + (lane/16)*8, col lane%16)
    const int m0 = mBlk + wmL;
    const int n0 = nBlk + wnL;
    #pragma unroll
    for (int ni = 0; ni < 4; ++ni) {
        const float bv = bias[n0 + ni * 16 + lm];
        #pragma unroll
        for (int mi = 0; mi < 4; ++mi) {
            #pragma unroll
            for (int r = 0; r < 8; ++r) {
                const int m = m0 + mi * 16 + ((lane >> 4) << 3) + r;
                C[(size_t)m * OUT_DIM + n0 + ni * 16 + lm] = acc[mi][ni][r] + bv;
            }
        }
    }
}

extern "C" void kernel_launch(void* const* d_in, const int* in_sizes, int n_in,
                              void* d_out, int out_size, void* d_ws, size_t ws_size,
                              hipStream_t stream) {
    const float* x   = (const float*)d_in[0];
    const float* wm  = (const float*)d_in[1];
    const float* wsp = (const float*)d_in[2];
    const float* bm  = (const float*)d_in[3];
    const float* bsp = (const float*)d_in[4];
    const float* nw  = (const float*)d_in[5];
    const float* nb  = (const float*)d_in[6];
    float* out = (float*)d_out;

    // Workspace layout: [Wb bf16 32MB][Xb bf16 8MB][bias f32 16KB]
    char* ws = (char*)d_ws;
    unsigned short* Wbuf = (unsigned short*)ws;
    unsigned short* Xbuf = (unsigned short*)(ws + (size_t)OUT_DIM * IN_DIM * 2);
    float* biasbuf = (float*)(ws + (size_t)OUT_DIM * IN_DIM * 2 + (size_t)B_ROWS * IN_DIM * 2);

    const int wblocks = (OUT_DIM * IN_DIM) / (256 * 4);   // 16384
    const int xblocks = (B_ROWS * IN_DIM) / (256 * 4);    // 4096
    mat_weight_kernel<<<wblocks, 256, 0, stream>>>(wm, wsp, nw, Wbuf);
    mat_x_kernel<<<xblocks, 256, 0, stream>>>(x, Xbuf);
    mat_bias_kernel<<<OUT_DIM / 256, 256, 0, stream>>>(bm, bsp, nb, biasbuf);

    dim3 grid(OUT_DIM / BN, B_ROWS / BM);                 // 16 x 8
    gemm_wmma_kernel<<<grid, 256, 0, stream>>>((const __bf16*)Xbuf, (const __bf16*)Wbuf,
                                               biasbuf, out);
}
